// Seq2seqLuongAttention_32538672235169
// MI455X (gfx1250) — matrix operand order repaired and checked by reading, not compiled, not run on hardware
//
#include <hip/hip_runtime.h>
#include <hip/hip_bf16.h>

// ---- problem sizes (match reference) ----
#define HH  512
#define EE  512
#define VV  32000
#define BB  64
#define TSS 256
#define TDD 64

typedef __attribute__((ext_vector_type(16))) __bf16 v16bf;
typedef __attribute__((ext_vector_type(8)))  float  v8f;

__device__ __forceinline__ float sigmoidf_(float x) { return 1.f / (1.f + __expf(-x)); }
__device__ __forceinline__ float tanhf_(float x)    { return 2.f * sigmoidf_(2.f * x) - 1.f; }

// A-fragment: 16x32 bf16, row-major source with leading dim lda (elements).
// ISA layout: lanes 0-15 -> rows, K in {0..7, 16..23}; lanes 16-31 -> rows, K in {8..15, 24..31}.
__device__ __forceinline__ v16bf load_a_frag(const __bf16* __restrict__ A, int lda) {
  int lane = threadIdx.x & 31;
  const __bf16* p = A + (size_t)(lane & 15) * lda + ((lane >> 4) << 3);
  union { v16bf f; float4 q[2]; } u;
  u.q[0] = *(const float4*)(p);
  u.q[1] = *(const float4*)(p + 16);
  return u.f;
}

// B-fragment: 32x16 bf16, row-major [K,N] source with leading dim ldb.
// ISA layout: lanes 0-15 -> cols N, K in {0..7, 16..23}; lanes 16-31 -> cols N, K in {8..15, 24..31}.
__device__ __forceinline__ v16bf load_b_frag(const __bf16* __restrict__ Bp, int ldb) {
  int lane = threadIdx.x & 31;
  const __bf16* p = Bp + (size_t)((lane >> 4) << 3) * ldb + (lane & 15);
  v16bf f;
#pragma unroll
  for (int s = 0; s < 8; ++s) {
    f[s] = p[(size_t)s * ldb];
    f[s + 8] = p[(size_t)(16 + s) * ldb];
  }
  return f;
}

// Device-wide spin barrier (all blocks of the launch must be co-resident; we use <=64 blocks).
// Monotonic counter in global memory; release on signal, acquire on observe.
__device__ __forceinline__ void grid_barrier(int* bar, int target) {
  __syncthreads();
  if (threadIdx.x == 0) {
    __hip_atomic_fetch_add(bar, 1, __ATOMIC_RELEASE, __HIP_MEMORY_SCOPE_AGENT);
    while (__hip_atomic_load(bar, __ATOMIC_ACQUIRE, __HIP_MEMORY_SCOPE_AGENT) < target)
      __builtin_amdgcn_s_sleep(1);
  }
  __syncthreads();
}

// ---------------- generic bf16 GEMM: C[M,N] = A[M,K] @ B[K,N] (+bias[N]) in fp32 ----------------
__global__ __launch_bounds__(256) void gemm_bf16_kernel(
    const __bf16* __restrict__ A, const __bf16* __restrict__ B,
    const float* __restrict__ bias, float* __restrict__ C,
    int M, int N, int K)
{
  int wid = (blockIdx.x * blockDim.x + threadIdx.x) >> 5;
  int nGroups = N >> 6;
  int total = (M >> 4) * nGroups;
  if (wid >= total) return;
  int mt = wid / nGroups;
  int ng = wid - mt * nGroups;

  v8f acc[4];
  for (int j = 0; j < 4; ++j)
    for (int i = 0; i < 8; ++i) acc[j][i] = 0.f;

  const __bf16* Arow = A + (size_t)mt * 16 * K;
  const __bf16* Bcol = B + (size_t)ng * 64;
  for (int k = 0; k < K; k += 32) {
    v16bf a = load_a_frag(Arow + k, K);
#pragma unroll
    for (int j = 0; j < 4; ++j) {
      v16bf b = load_b_frag(Bcol + (size_t)k * N + j * 16, N);
      acc[j] = __builtin_amdgcn_wmma_f32_16x16x32_bf16(false, a, false, b,
                                                       (short)0, acc[j], false, false);
    }
  }

  int lane = threadIdx.x & 31;
  int n0 = lane & 15;
  int mh = (lane >> 4) << 3;
  for (int j = 0; j < 4; ++j) {
    int col = ng * 64 + j * 16 + n0;
    float bv = bias ? bias[col] : 0.f;
#pragma unroll
    for (int v = 0; v < 8; ++v) {
      int row = mt * 16 + mh + v;
      C[(size_t)row * N + col] = acc[j][v] + bv;
    }
  }
}

// ---------------- persistent encoder: all 256 GRU steps in one launch ----------------
// 32 blocks x 128 threads = 128 waves; wave owns a 16(batch) x 16(H) state tile and
// computes rec_{z,r,h} (3 WMMA tile-GEMMs, K=512) + gate math per step; grid barrier between steps.
__global__ __launch_bounds__(128) void enc_persist_kernel(
    float* __restrict__ h0f, __bf16* __restrict__ h0b,
    float* __restrict__ h1f, __bf16* __restrict__ h1b,
    const float* __restrict__ gx,      // [B*TS, 3H] (enc_b folded in)
    const __bf16* __restrict__ Wh,     // [H, 3H]
    __bf16* __restrict__ enc_hs,       // [B, TS, H] bf16
    int* __restrict__ bar)
{
  int wid = blockIdx.x * 4 + (threadIdx.x >> 5);
  int bt = wid >> 5;   // 0..3
  int jt = wid & 31;   // 0..31
  int lane = threadIdx.x & 31;
  int n0 = lane & 15;
  int mh = (lane >> 4) << 3;
  int j = jt * 16 + n0;

  for (int t = 0; t < TSS; ++t) {
    const float*  h_in   = (t & 1) ? h1f : h0f;
    const __bf16* hbf_in = (t & 1) ? h1b : h0b;
    float*  h_out   = (t & 1) ? h0f : h1f;
    __bf16* hbf_out = (t & 1) ? h0b : h1b;

    v8f rec[3];
    for (int g = 0; g < 3; ++g)
      for (int i = 0; i < 8; ++i) rec[g][i] = 0.f;

    const __bf16* Arow = hbf_in + (size_t)bt * 16 * HH;
    for (int k = 0; k < HH; k += 32) {
      v16bf a = load_a_frag(Arow + k, HH);
#pragma unroll
      for (int g = 0; g < 3; ++g) {
        v16bf b = load_b_frag(Wh + (size_t)k * (3 * HH) + g * HH + jt * 16, 3 * HH);
        rec[g] = __builtin_amdgcn_wmma_f32_16x16x32_bf16(false, a, false, b,
                                                         (short)0, rec[g], false, false);
      }
    }

#pragma unroll
    for (int v = 0; v < 8; ++v) {
      int b = bt * 16 + mh + v;
      const float* g = gx + ((size_t)b * TSS + t) * (3 * HH);
      float z  = sigmoidf_(g[j] + rec[0][v]);
      float r  = sigmoidf_(g[HH + j] + rec[1][v]);
      float hh = tanhf_(g[2 * HH + j] + r * rec[2][v]);
      float h2 = z * h_in[b * HH + j] + (1.f - z) * hh;
      h_out[b * HH + j] = h2;
      __bf16 hb = (__bf16)h2;
      hbf_out[b * HH + j] = hb;
      enc_hs[((size_t)b * TSS + t) * HH + j] = hb;
    }

    grid_barrier(bar, gridDim.x * (t + 1));
  }
}

// ---------------- persistent decoder: all 64 attention+GRU steps in one launch ----------------
// 64 blocks x 256 threads = 512 waves. Per step:
//   phase1: q = h @ W_att (128 WMMA tiles over the waves)
//   phase2: block b does softmax attention for batch b, builds inp=[x_t | ctx]
//   phase3: fused GRU step (gx K=1024 + rec K=512 WMMA tiles + gates)
__global__ __launch_bounds__(256) void dec_persist_kernel(
    float* __restrict__ h0f, __bf16* __restrict__ h0b,
    float* __restrict__ h1f, __bf16* __restrict__ h1b,
    const __bf16* __restrict__ watt,   // [H, H]
    float* __restrict__ qbuf,          // [B, H]
    const __bf16* __restrict__ enc_hs, // [B, TS, H]
    const __bf16* __restrict__ decE,   // [B*TD, E]
    __bf16* __restrict__ inp,          // [B, E+H]
    const __bf16* __restrict__ dWx,    // [E+H, 3H]
    const float* __restrict__ db,      // [3H]
    const __bf16* __restrict__ dWh,    // [H, 3H]
    __bf16* __restrict__ outs,         // [B, TD, H]
    int* __restrict__ bar)
{
  __shared__ float qs[HH];
  __shared__ float sc[TSS];
  __shared__ float tmp[TSS];
  const int KX = EE + HH;

  int tid = threadIdx.x;
  int gw = blockIdx.x * 8 + (tid >> 5);  // 0..511 global wave id
  int lane = tid & 31;
  int n0 = lane & 15;
  int mh = (lane >> 4) << 3;
  int target = 0;

  for (int t = 0; t < TDD; ++t) {
    const float*  h_in   = (t & 1) ? h1f : h0f;
    const __bf16* hbf_in = (t & 1) ? h1b : h0b;
    float*  h_out   = (t & 1) ? h0f : h1f;
    __bf16* hbf_out = (t & 1) ? h0b : h1b;

    // ---- phase 1: q = h @ W_att ----
    if (gw < 128) {
      int mt = gw >> 5, nt = gw & 31;
      v8f acc;
      for (int i = 0; i < 8; ++i) acc[i] = 0.f;
      const __bf16* Arow = hbf_in + (size_t)mt * 16 * HH;
      for (int k = 0; k < HH; k += 32) {
        v16bf a = load_a_frag(Arow + k, HH);
        v16bf b = load_b_frag(watt + (size_t)k * HH + nt * 16, HH);
        acc = __builtin_amdgcn_wmma_f32_16x16x32_bf16(false, a, false, b,
                                                      (short)0, acc, false, false);
      }
#pragma unroll
      for (int v = 0; v < 8; ++v)
        qbuf[(size_t)(mt * 16 + mh + v) * HH + nt * 16 + n0] = acc[v];
    }
    target += gridDim.x;
    grid_barrier(bar, target);

    // ---- phase 2: attention for batch row b = blockIdx.x ----
    {
      int b = blockIdx.x;
      qs[tid] = qbuf[b * HH + tid];
      qs[tid + 256] = qbuf[b * HH + tid + 256];
      __syncthreads();

      const __bf16* row = enc_hs + ((size_t)b * TSS + tid) * HH;
      float s = 0.f;
#pragma unroll 8
      for (int h = 0; h < HH; ++h) s += (float)row[h] * qs[h];
      tmp[tid] = s;
      __syncthreads();
      for (int off = 128; off > 0; off >>= 1) {
        if (tid < off) tmp[tid] = fmaxf(tmp[tid], tmp[tid + off]);
        __syncthreads();
      }
      float m = tmp[0];
      __syncthreads();
      float e = __expf(s - m);
      tmp[tid] = e;
      __syncthreads();
      for (int off = 128; off > 0; off >>= 1) {
        if (tid < off) tmp[tid] += tmp[tid + off];
        __syncthreads();
      }
      float inv = 1.f / tmp[0];
      sc[tid] = e * inv;
      __syncthreads();

      float a0 = 0.f, a1 = 0.f;
      for (int ts = 0; ts < TSS; ++ts) {
        float w = sc[ts];
        const __bf16* r2 = enc_hs + ((size_t)b * TSS + ts) * HH;
        a0 += w * (float)r2[tid];
        a1 += w * (float)r2[tid + 256];
      }
      const __bf16* x = decE + ((size_t)b * TDD + t) * EE;
      __bf16* dst = inp + (size_t)b * KX;
      dst[tid] = x[tid];
      dst[tid + 256] = x[tid + 256];
      dst[EE + tid] = (__bf16)a0;
      dst[EE + tid + 256] = (__bf16)a1;
    }
    target += gridDim.x;
    grid_barrier(bar, target);

    // ---- phase 3: fused GRU step ----
    if (gw < 128) {
      int bt = gw >> 5, jt = gw & 31;
      int j = jt * 16 + n0;

      v8f gxa[3], rec[3];
      for (int g = 0; g < 3; ++g)
        for (int i = 0; i < 8; ++i) { gxa[g][i] = 0.f; rec[g][i] = 0.f; }

      const __bf16* Ax = inp + (size_t)bt * 16 * KX;
      for (int k = 0; k < KX; k += 32) {
        v16bf a = load_a_frag(Ax + k, KX);
#pragma unroll
        for (int g = 0; g < 3; ++g) {
          v16bf b = load_b_frag(dWx + (size_t)k * (3 * HH) + g * HH + jt * 16, 3 * HH);
          gxa[g] = __builtin_amdgcn_wmma_f32_16x16x32_bf16(false, a, false, b,
                                                           (short)0, gxa[g], false, false);
        }
      }
      const __bf16* Ah = hbf_in + (size_t)bt * 16 * HH;
      for (int k = 0; k < HH; k += 32) {
        v16bf a = load_a_frag(Ah + k, HH);
#pragma unroll
        for (int g = 0; g < 3; ++g) {
          v16bf b = load_b_frag(dWh + (size_t)k * (3 * HH) + g * HH + jt * 16, 3 * HH);
          rec[g] = __builtin_amdgcn_wmma_f32_16x16x32_bf16(false, a, false, b,
                                                           (short)0, rec[g], false, false);
        }
      }

      float b0 = db[j], b1 = db[HH + j], b2 = db[2 * HH + j];
#pragma unroll
      for (int v = 0; v < 8; ++v) {
        int b = bt * 16 + mh + v;
        float z  = sigmoidf_(gxa[0][v] + b0 + rec[0][v]);
        float r  = sigmoidf_(gxa[1][v] + b1 + rec[1][v]);
        float hh = tanhf_(gxa[2][v] + b2 + r * rec[2][v]);
        float h2 = z * h_in[b * HH + j] + (1.f - z) * hh;
        h_out[b * HH + j] = h2;
        __bf16 hb = (__bf16)h2;
        hbf_out[b * HH + j] = hb;
        outs[((size_t)b * TDD + t) * HH + j] = hb;
      }
    }
    target += gridDim.x;
    grid_barrier(bar, target);
  }
}

// ---------------- utilities ----------------
__global__ void cvt_f32_bf16_kernel(const float* __restrict__ s, __bf16* __restrict__ d, int n) {
  int i = blockIdx.x * blockDim.x + threadIdx.x;
  int stride = gridDim.x * blockDim.x;
  for (; i < n; i += stride) d[i] = (__bf16)s[i];
}

__global__ void gather_embed_kernel(const int* __restrict__ tok, int ntok,
                                    const float* __restrict__ emb, __bf16* __restrict__ d) {
  int n = ntok * EE;
  int i = blockIdx.x * blockDim.x + threadIdx.x;
  int stride = gridDim.x * blockDim.x;
  for (; i < n; i += stride) {
    int r = i >> 9;          // / EE
    int e = i & (EE - 1);
    d[i] = (__bf16)emb[(size_t)tok[r] * EE + e];
  }
}

// zero h state and both barrier counters (must be reset every call: graph replays don't re-poison)
__global__ void zero_init_kernel(float* h, __bf16* hb, int* bars) {
  int i = blockIdx.x * blockDim.x + threadIdx.x;
  if (i < BB * HH) { h[i] = 0.f; hb[i] = (__bf16)0.f; }
  if (i == 0) { bars[0] = 0; bars[1] = 0; }
}

// ---------------- launcher ----------------
extern "C" void kernel_launch(void* const* d_in, const int* in_sizes, int n_in,
                              void* d_out, int out_size, void* d_ws, size_t ws_size,
                              hipStream_t stream) {
  const int*   src_seq = (const int*)d_in[0];
  const int*   dec_seq = (const int*)d_in[1];
  const float* emb     = (const float*)d_in[2];
  const float* enc_Wx  = (const float*)d_in[3];
  const float* enc_Wh  = (const float*)d_in[4];
  const float* enc_b   = (const float*)d_in[5];
  const float* dec_Wx  = (const float*)d_in[6];
  const float* dec_Wh  = (const float*)d_in[7];
  const float* dec_b   = (const float*)d_in[8];
  const float* W_att   = (const float*)d_in[9];
  const float* Wd      = (const float*)d_in[10];
  const float* bd      = (const float*)d_in[11];
  float* out = (float*)d_out;

  char* base = (char*)d_ws;
  size_t off = 0;
  auto alloc = [&](size_t bytes) -> char* {
    char* p = base + off;
    off = (off + bytes + 255) & ~(size_t)255;
    return p;
  };

  __bf16* wx    = (__bf16*)alloc((size_t)EE * 3 * HH * 2);
  __bf16* wh    = (__bf16*)alloc((size_t)HH * 3 * HH * 2);
  __bf16* dwx   = (__bf16*)alloc((size_t)(EE + HH) * 3 * HH * 2);
  __bf16* dwh   = (__bf16*)alloc((size_t)HH * 3 * HH * 2);
  __bf16* watt  = (__bf16*)alloc((size_t)HH * HH * 2);
  __bf16* wd    = (__bf16*)alloc((size_t)HH * VV * 2);
  __bf16* srcE  = (__bf16*)alloc((size_t)BB * TSS * EE * 2);
  __bf16* decE  = (__bf16*)alloc((size_t)BB * TDD * EE * 2);
  float*  gxbuf = (float*)alloc((size_t)BB * TSS * 3 * HH * 4);
  float*  hf0   = (float*)alloc((size_t)BB * HH * 4);
  float*  hf1   = (float*)alloc((size_t)BB * HH * 4);
  __bf16* hb0   = (__bf16*)alloc((size_t)BB * HH * 2);
  __bf16* hb1   = (__bf16*)alloc((size_t)BB * HH * 2);
  __bf16* enchs = (__bf16*)alloc((size_t)BB * TSS * HH * 2);
  __bf16* outsb = (__bf16*)alloc((size_t)BB * TDD * HH * 2);
  float*  qbuf  = (float*)alloc((size_t)BB * HH * 4);
  __bf16* inpb  = (__bf16*)alloc((size_t)BB * (EE + HH) * 2);
  int*    bars  = (int*)alloc(256);

  auto launch_gemm = [&](const __bf16* A, const __bf16* B, const float* bias,
                         float* C, int M, int N, int K) {
    int waves = (M / 16) * (N / 64);
    int blocks = (waves + 7) / 8;
    gemm_bf16_kernel<<<blocks, 256, 0, stream>>>(A, B, bias, C, M, N, K);
  };

  // --- setup: bf16 weights, embeddings, zeroed state + barrier counters ---
  cvt_f32_bf16_kernel<<<512, 256, 0, stream>>>(enc_Wx, wx, EE * 3 * HH);
  cvt_f32_bf16_kernel<<<512, 256, 0, stream>>>(enc_Wh, wh, HH * 3 * HH);
  cvt_f32_bf16_kernel<<<512, 256, 0, stream>>>(dec_Wx, dwx, (EE + HH) * 3 * HH);
  cvt_f32_bf16_kernel<<<512, 256, 0, stream>>>(dec_Wh, dwh, HH * 3 * HH);
  cvt_f32_bf16_kernel<<<256, 256, 0, stream>>>(W_att, watt, HH * HH);
  cvt_f32_bf16_kernel<<<2048, 256, 0, stream>>>(Wd, wd, HH * VV);
  gather_embed_kernel<<<2048, 256, 0, stream>>>(src_seq, BB * TSS, emb, srcE);
  gather_embed_kernel<<<1024, 256, 0, stream>>>(dec_seq, BB * TDD, emb, decE);
  zero_init_kernel<<<(BB * HH + 255) / 256, 256, 0, stream>>>(hf0, hb0, bars);

  // --- hoisted encoder input GEMM: gx = src_e @ enc_Wx + enc_b  [16384 x 1536] ---
  launch_gemm(srcE, wx, enc_b, gxbuf, BB * TSS, 3 * HH, EE);

  // --- encoder recurrence: ONE persistent launch, 256 steps with grid barriers ---
  enc_persist_kernel<<<32, 128, 0, stream>>>(hf0, hb0, hf1, hb1, gxbuf, wh, enchs, bars + 0);

  // --- decoder: ONE persistent launch, 64 steps x (q GEMM | attention | GRU) ---
  // encoder ends with h_last in parity 0 (256 even), matching decoder's t=0 read parity.
  dec_persist_kernel<<<64, 256, 0, stream>>>(hf0, hb0, hf1, hb1, watt, qbuf,
                                             enchs, decE, inpb, dwx, dec_b, dwh,
                                             outsb, bars + 1);

  // --- vocab projection: logits = outs @ Wd + bd  [4096 x 32000] ---
  launch_gemm(outsb, wd, bd, out, BB * TDD, VV, HH);
}
